// OptimizedRWKVBlock_824633721103
// MI455X (gfx1250) — matrix operand order repaired and checked by reading, not compiled, not run on hardware
//
#include <hip/hip_runtime.h>
#include <hip/hip_bf16.h>
#include <cstdint>

typedef __bf16 bf16;
typedef __attribute__((ext_vector_type(16))) __bf16 v16bf;
typedef __attribute__((ext_vector_type(8)))  float  v8f;
typedef __attribute__((ext_vector_type(4)))  int    v4i;

#define LN_EPS   1e-5f
#define WKV_EPS  1e-8f
#define NEG_INF  -1e38f

static constexpr int BB = 8;
static constexpr int TT = 2048;
static constexpr int CC = 1024;
static constexpr long MM = (long)BB * TT;   // 16384 rows

// ---------------------------------------------------------------------------
// CDNA5 async global->LDS copy (ASYNCcnt-tracked), with safe fallback.
// Builtin prototype (from hipcc diagnostic): arg0 is int4* in global AS.
// ---------------------------------------------------------------------------
#if defined(__HIP_DEVICE_COMPILE__) && \
    __has_builtin(__builtin_amdgcn_global_load_async_to_lds_b128)
#define USE_ASYNC_LDS 1
#else
#define USE_ASYNC_LDS 0
#endif

typedef __attribute__((address_space(1))) v4i gv4i;
typedef __attribute__((address_space(3))) v4i lv4i;

__device__ __forceinline__ void cp_async16(const bf16* g, bf16* l) {
#if USE_ASYNC_LDS
    __builtin_amdgcn_global_load_async_to_lds_b128(
        (gv4i*)(uintptr_t)g,
        (lv4i*)(uint32_t)(uintptr_t)l,
        0, 0);
#else
    (void)g; (void)l;
#endif
}

__device__ __forceinline__ void wait_async0() {
#if __has_builtin(__builtin_amdgcn_s_wait_asynccnt)
    __builtin_amdgcn_s_wait_asynccnt(0);
#else
    asm volatile("s_wait_asynccnt 0" ::: "memory");
#endif
}

// ---------------------------------------------------------------------------
// Weight convert + transpose: W (K x N, fp32 row-major) -> Wt (N x K, bf16)
// ---------------------------------------------------------------------------
__global__ __launch_bounds__(256) void k_convert_wT(const float* __restrict__ W,
                                                    bf16* __restrict__ Wt,
                                                    int N, int kshift) {
    long i = (long)blockIdx.x * 256 + threadIdx.x;
    int K = 1 << kshift;
    long total = (long)N * K;
    if (i >= total) return;
    int n = (int)(i >> kshift);
    int k = (int)(i & (K - 1));
    Wt[i] = (bf16)W[(long)k * N + n];
}

// ---------------------------------------------------------------------------
// LayerNorm 1: xn = LN(x; g, b) -> bf16
// ---------------------------------------------------------------------------
__global__ __launch_bounds__(256) void k_ln1(const float* __restrict__ x,
                                             const float* __restrict__ g,
                                             const float* __restrict__ b,
                                             bf16* __restrict__ xn) {
    long row = blockIdx.x;
    const float* xr = x + row * CC;
    int tid = threadIdx.x;
    float vals[4];
    float s = 0.f, s2 = 0.f;
    #pragma unroll
    for (int i = 0; i < 4; ++i) {
        float v = xr[tid + i * 256];
        vals[i] = v; s += v; s2 += v * v;
    }
    __shared__ float red1[256], red2[256];
    red1[tid] = s; red2[tid] = s2;
    __syncthreads();
    for (int off = 128; off > 0; off >>= 1) {
        if (tid < off) { red1[tid] += red1[tid + off]; red2[tid] += red2[tid + off]; }
        __syncthreads();
    }
    float mean = red1[0] * (1.f / CC);
    float var  = red2[0] * (1.f / CC) - mean * mean;
    float inv  = rsqrtf(var + LN_EPS);
    #pragma unroll
    for (int i = 0; i < 4; ++i) {
        int c = tid + i * 256;
        xn[row * CC + c] = (bf16)((vals[i] - mean) * inv * g[c] + b[c]);
    }
}

// ---------------------------------------------------------------------------
// LayerNorm 2 + receptance gate
// ---------------------------------------------------------------------------
__global__ __launch_bounds__(256) void k_ln2(const float* __restrict__ x2,
                                             const float* __restrict__ g,
                                             const float* __restrict__ b,
                                             const float* __restrict__ cmr,
                                             bf16* __restrict__ xn,
                                             float* __restrict__ rr) {
    long row = blockIdx.x;
    int bidx = (int)(row >> 11);
    int t    = (int)(row & (TT - 1));
    long prow = (long)bidx * TT + (t == 0 ? TT - 1 : t - 1);
    const float* xr = x2 + row * CC;
    const float* xp = x2 + prow * CC;
    int tid = threadIdx.x;
    float vals[4];
    float s = 0.f, s2 = 0.f;
    #pragma unroll
    for (int i = 0; i < 4; ++i) {
        float v = xr[tid + i * 256];
        vals[i] = v; s += v; s2 += v * v;
    }
    __shared__ float red1[256], red2[256];
    red1[tid] = s; red2[tid] = s2;
    __syncthreads();
    for (int off = 128; off > 0; off >>= 1) {
        if (tid < off) { red1[tid] += red1[tid + off]; red2[tid] += red2[tid + off]; }
        __syncthreads();
    }
    float mean = red1[0] * (1.f / CC);
    float var  = red2[0] * (1.f / CC) - mean * mean;
    float inv  = rsqrtf(var + LN_EPS);
    #pragma unroll
    for (int i = 0; i < 4; ++i) {
        int c = tid + i * 256;
        float xnv = (vals[i] - mean) * inv * g[c] + b[c];
        xn[row * CC + c] = (bf16)xnv;
        float m = cmr[c];
        float z = xnv * m + xp[c] * (1.f - m);
        rr[row * CC + c] = 1.f / (1.f + __expf(-z));
    }
}

// ---------------------------------------------------------------------------
// WKV: fused token-shift mixing + sigmoid(r) + log-sum-exp recurrence.
// ---------------------------------------------------------------------------
__global__ __launch_bounds__(256) void k_wkv(const float* __restrict__ x,
                                             const float* __restrict__ kvr,
                                             const float* __restrict__ wdec,
                                             const float* __restrict__ ufirst,
                                             const float* __restrict__ tmk,
                                             const float* __restrict__ tmv,
                                             const float* __restrict__ tmr,
                                             float* __restrict__ xout,
                                             float* __restrict__ state) {
    int idx = blockIdx.x * 256 + threadIdx.x;     // B*C = 8192 lanes
    int bidx = idx >> 10;
    int c    = idx & (CC - 1);
    float w = wdec[c], u = ufirst[c];
    float mk = tmk[c], mv = tmv[c], mr = tmr[c];
    float aa = 0.f, bb = NEG_INF;
    long rbase = (long)bidx * TT;
    float xprev = x[rbase * CC + c];
    for (int t = 0; t < TT; ++t) {
        long ro = rbase + t;
        float xc = x[ro * CC + c];
        const float* kr = kvr + ro * (3 * CC);
        float kk = kr[c]          * mk + xprev * (1.f - mk);
        float vv = kr[CC + c]     * mv + xprev * (1.f - mv);
        float rm = kr[2 * CC + c] * mr + xprev * (1.f - mr);
        float r  = 1.f / (1.f + __expf(-rm));
        float ww = u + kk;
        float p  = fmaxf(bb, ww);
        float e1 = __expf(bb - p);
        float e2 = __expf(ww - p);
        float y  = (e1 * aa + e2 * vv) / (e1 + e2 + WKV_EPS);
        float w2 = w + bb;
        float p2 = fmaxf(w2, kk);
        float f1 = __expf(w2 - p2);
        float f2 = __expf(kk - p2);
        aa = f1 * aa + f2 * vv;
        bb = p2 + __logf(f1 + f2 + WKV_EPS);
        xout[ro * CC + c] = xc + r * y;
        xprev = xc;
    }
    state[(long)(bidx * CC + c) * 2 + 0] = aa;
    state[(long)(bidx * CC + c) * 2 + 1] = bb;
}

// ---------------------------------------------------------------------------
// bf16 WMMA GEMM with double-buffered LDS + pipelined (async) staging.
//   C(M x N) = A(M x K) * Bt(N x K)^T, 128x128 tile, 8 waves, 2x4 WMMA/wave.
// MODE 0: store fp32   MODE 1: mix+relu^2 -> bf16 hid   MODE 2: xout += rr*c
// ---------------------------------------------------------------------------
template <int MODE>
__global__ __launch_bounds__(256) void k_gemm(const bf16* __restrict__ A,
                                              const bf16* __restrict__ Bt,
                                              float* __restrict__ Cout,
                                              int K, int N,
                                              const float* __restrict__ cmk,
                                              const float* __restrict__ xsrc,
                                              const float* __restrict__ rr,
                                              bf16* __restrict__ hid,
                                              float* __restrict__ xout) {
    __shared__ bf16 lA[2][128 * 32];
    __shared__ bf16 lB[2][128 * 32];

    const int tid  = threadIdx.x;
    const int lane = tid & 31;
    const int wid  = tid >> 5;
    const int l15  = lane & 15;
    const int lh   = lane >> 4;
    const int wrow = wid >> 1;            // 0..3 -> 32-row strip
    const int wcol = wid & 1;             // 0..1 -> 64-col strip
    const long m0 = (long)blockIdx.x * 128;
    const long n0 = (long)blockIdx.y * 128;

    // staging geometry: each thread moves 2 x 16B per matrix per K-step
    const int  srow  = tid >> 2;          // 0..63
    const int  scol  = (tid & 3) * 8;     // 0,8,16,24
    const bf16* gA = A  + (m0 + srow) * K + scol;
    const bf16* gB = Bt + (n0 + srow) * K + scol;
    const long rstep = (long)64 * K;      // chunk 1: +64 rows
    const int  lso   = srow * 32 + scol;  // LDS offset; +64*32 for chunk 1

    v8f acc[2][4];
    #pragma unroll
    for (int i = 0; i < 2; ++i)
        #pragma unroll
        for (int j = 0; j < 4; ++j)
            acc[i][j] = (v8f){0.f, 0.f, 0.f, 0.f, 0.f, 0.f, 0.f, 0.f};

    // fragment LDS offsets (contiguous k = 16*lh + s, same order for A and B)
    int aoff[2], boff[4];
    #pragma unroll
    for (int mt = 0; mt < 2; ++mt)
        aoff[mt] = (wrow * 32 + mt * 16 + l15) * 32 + lh * 16;
    #pragma unroll
    for (int nt = 0; nt < 4; ++nt)
        boff[nt] = (wcol * 64 + nt * 16 + l15) * 32 + lh * 16;

    const int nk = K >> 5;
    union Frag { v16bf v; uint4 u[2]; };

#if USE_ASYNC_LDS
    // -------- async global->LDS pipeline --------
    {
        #pragma unroll
        for (int it = 0; it < 2; ++it) {
            cp_async16(gA + it * rstep, &lA[0][lso + it * 64 * 32]);
            cp_async16(gB + it * rstep, &lB[0][lso + it * 64 * 32]);
        }
    }
    int buf = 0;
    for (int ki = 0; ki < nk; ++ki) {
        wait_async0();
        __syncthreads();
        if (ki + 1 < nk) {
            int ko = (ki + 1) * 32;
            #pragma unroll
            for (int it = 0; it < 2; ++it) {
                cp_async16(gA + ko + it * rstep, &lA[buf ^ 1][lso + it * 64 * 32]);
                cp_async16(gB + ko + it * rstep, &lB[buf ^ 1][lso + it * 64 * 32]);
            }
        }
        const bf16* la = lA[buf];
        const bf16* lb = lB[buf];
        Frag fa[2], fb[4];
        #pragma unroll
        for (int mt = 0; mt < 2; ++mt) {
            fa[mt].u[0] = *(const uint4*)&la[aoff[mt]];
            fa[mt].u[1] = *(const uint4*)&la[aoff[mt] + 8];
        }
        #pragma unroll
        for (int nt = 0; nt < 4; ++nt) {
            fb[nt].u[0] = *(const uint4*)&lb[boff[nt]];
            fb[nt].u[1] = *(const uint4*)&lb[boff[nt] + 8];
        }
        #pragma unroll
        for (int mt = 0; mt < 2; ++mt)
            #pragma unroll
            for (int nt = 0; nt < 4; ++nt)
                acc[mt][nt] = __builtin_amdgcn_wmma_f32_16x16x32_bf16(
                    false, fa[mt].v, false, fb[nt].v,
                    (short)0, acc[mt][nt], false, false);
        buf ^= 1;
    }
#else
    // -------- register-pipelined fallback: issue all loads, wait once --------
    uint4 ra[2], rb[2];
    {
        #pragma unroll
        for (int it = 0; it < 2; ++it) {
            ra[it] = *(const uint4*)(gA + it * rstep);
            rb[it] = *(const uint4*)(gB + it * rstep);
        }
    }
    int buf = 0;
    for (int ki = 0; ki < nk; ++ki) {
        #pragma unroll
        for (int it = 0; it < 2; ++it) {
            *(uint4*)&lA[buf][lso + it * 64 * 32] = ra[it];
            *(uint4*)&lB[buf][lso + it * 64 * 32] = rb[it];
        }
        __syncthreads();
        if (ki + 1 < nk) {
            int ko = (ki + 1) * 32;
            #pragma unroll
            for (int it = 0; it < 2; ++it) {
                ra[it] = *(const uint4*)(gA + ko + it * rstep);
                rb[it] = *(const uint4*)(gB + ko + it * rstep);
            }
        }
        const bf16* la = lA[buf];
        const bf16* lb = lB[buf];
        Frag fa[2], fb[4];
        #pragma unroll
        for (int mt = 0; mt < 2; ++mt) {
            fa[mt].u[0] = *(const uint4*)&la[aoff[mt]];
            fa[mt].u[1] = *(const uint4*)&la[aoff[mt] + 8];
        }
        #pragma unroll
        for (int nt = 0; nt < 4; ++nt) {
            fb[nt].u[0] = *(const uint4*)&lb[boff[nt]];
            fb[nt].u[1] = *(const uint4*)&lb[boff[nt] + 8];
        }
        #pragma unroll
        for (int mt = 0; mt < 2; ++mt)
            #pragma unroll
            for (int nt = 0; nt < 4; ++nt)
                acc[mt][nt] = __builtin_amdgcn_wmma_f32_16x16x32_bf16(
                    false, fa[mt].v, false, fb[nt].v,
                    (short)0, acc[mt][nt], false, false);
        buf ^= 1;
    }
#endif

    // ---- epilogue (C layout: VGPR i -> row (lh*8 + i), col = l15) ----
    #pragma unroll
    for (int mt = 0; mt < 2; ++mt) {
        #pragma unroll
        for (int nt = 0; nt < 4; ++nt) {
            long rbase = m0 + wrow * 32 + mt * 16 + lh * 8;
            int  col   = (int)n0 + wcol * 64 + nt * 16 + l15;
            #pragma unroll
            for (int i = 0; i < 8; ++i) {
                long row = rbase + i;
                float v = acc[mt][nt][i];
                if (MODE == 0) {
                    Cout[row * N + col] = v;
                } else if (MODE == 1) {
                    int cidx = col & (CC - 1);
                    float mkv = cmk[cidx];
                    int bidx = (int)(row >> 11);
                    int t    = (int)(row & (TT - 1));
                    long prow = (long)bidx * TT + (t == 0 ? TT - 1 : t - 1);
                    float xpv = xsrc[prow * CC + cidx];
                    float km  = v * mkv + xpv * (1.f - mkv);
                    km = fmaxf(km, 0.f);
                    hid[row * (long)N + col] = (bf16)(km * km);
                } else {
                    long o = row * (long)N + col;   // N == C
                    xout[o] = xout[o] + rr[o] * v;
                }
            }
        }
    }
}

// ---------------------------------------------------------------------------
extern "C" void kernel_launch(void* const* d_in, const int* in_sizes, int n_in,
                              void* d_out, int out_size, void* d_ws, size_t ws_size,
                              hipStream_t stream) {
    const float* x          = (const float*)d_in[0];
    const float* time_decay = (const float*)d_in[1];
    const float* time_first = (const float*)d_in[2];
    const float* W_tm       = (const float*)d_in[3];   // C x 3C
    const float* g1         = (const float*)d_in[4];
    const float* b1         = (const float*)d_in[5];
    const float* tmk        = (const float*)d_in[6];
    const float* tmv        = (const float*)d_in[7];
    const float* tmr        = (const float*)d_in[8];
    const float* W_cm       = (const float*)d_in[9];   // C x 4C
    const float* W_cp       = (const float*)d_in[10];  // 4C x C
    const float* g2         = (const float*)d_in[11];
    const float* b2         = (const float*)d_in[12];
    const float* cmk        = (const float*)d_in[13];
    const float* cmr        = (const float*)d_in[14];

    float* xout  = (float*)d_out;                 // M*C fp32
    float* state = xout + (size_t)MM * CC;        // B*C*2 fp32

    char* ws = (char*)d_ws;
    auto take = [&](size_t bytes) { char* p = ws; ws += (bytes + 255) & ~(size_t)255; return p; };
    bf16*  Wt_tm = (bf16*)take((size_t)3 * CC * CC * sizeof(bf16));   // (3C x C)
    bf16*  Wt_cm = (bf16*)take((size_t)4 * CC * CC * sizeof(bf16));   // (4C x C)
    bf16*  Wt_cp = (bf16*)take((size_t)CC * 4 * CC * sizeof(bf16));   // (C x 4C)
    bf16*  xn    = (bf16*)take((size_t)MM * CC * sizeof(bf16));
    float* kvr   = (float*)take((size_t)MM * 3 * CC * sizeof(float));
    bf16*  hid   = (bf16*)take((size_t)MM * 4 * CC * sizeof(bf16));
    float* rrbuf = (float*)take((size_t)MM * CC * sizeof(float));

    dim3 blk(256);

    k_convert_wT<<<(unsigned)((3L * CC * CC + 255) / 256), blk, 0, stream>>>(W_tm, Wt_tm, 3 * CC, 10);
    k_convert_wT<<<(unsigned)((4L * CC * CC + 255) / 256), blk, 0, stream>>>(W_cm, Wt_cm, 4 * CC, 10);
    k_convert_wT<<<(unsigned)((4L * CC * CC + 255) / 256), blk, 0, stream>>>(W_cp, Wt_cp, CC, 12);

    k_ln1<<<(unsigned)MM, blk, 0, stream>>>(x, g1, b1, xn);

    k_gemm<0><<<dim3(128, 24), blk, 0, stream>>>(xn, Wt_tm, kvr, CC, 3 * CC,
                                                 nullptr, nullptr, nullptr, nullptr, nullptr);

    k_wkv<<<(unsigned)(BB * CC / 256), blk, 0, stream>>>(x, kvr, time_decay, time_first,
                                                         tmk, tmv, tmr, xout, state);

    k_ln2<<<(unsigned)MM, blk, 0, stream>>>(xout, g2, b2, cmr, xn, rrbuf);

    k_gemm<1><<<dim3(128, 32), blk, 0, stream>>>(xn, Wt_cm, nullptr, CC, 4 * CC,
                                                 cmk, xout, nullptr, hid, nullptr);

    k_gemm<2><<<dim3(128, 8), blk, 0, stream>>>(hid, Wt_cp, nullptr, 4 * CC, CC,
                                                nullptr, nullptr, rrbuf, nullptr, xout);
}
